// MEGNetBlock_47974784696350
// MI455X (gfx1250) — compile-verified, matrix operand order checked by reading, not run on hardware
//
#include <hip/hip_runtime.h>
#include <hip/hip_bf16.h>

typedef __attribute__((ext_vector_type(16))) __bf16 v16bf;
typedef __attribute__((ext_vector_type(8)))  float  v8f;
typedef __attribute__((ext_vector_type(4)))  unsigned int v4u;
typedef __attribute__((ext_vector_type(8)))  int v8i_;
typedef __attribute__((ext_vector_type(4)))  int v4i_;

#define UDIM 128
#define NN 50000
#define EE 400000
#define GG 256

#if __has_builtin(__builtin_amdgcn_tensor_load_to_lds)
#define HAVE_TDM 1
#endif

// ---- workspace layout (bytes) ----
static const size_t OFF_WE1 = 0;                                   // [128][512] bf16
static const size_t OFF_WE2 = OFF_WE1 + (size_t)512 * 128 * 2;     // [128][128] bf16
static const size_t OFF_WN1 = OFF_WE2 + (size_t)128 * 128 * 2;     // [128][384] bf16
static const size_t OFF_WN2 = OFF_WN1 + (size_t)384 * 128 * 2;     // [128][128] bf16
static const size_t OFF_WG1 = OFF_WN2 + (size_t)128 * 128 * 2;     // [128][384] bf16
static const size_t OFF_WG2 = OFF_WG1 + (size_t)384 * 128 * 2;     // [128][128] bf16
static const size_t OFF_NODE_SUM = OFF_WG2 + (size_t)128 * 128 * 2;      // N*128 f32
static const size_t OFF_NODE_CNT = OFF_NODE_SUM + (size_t)NN * UDIM * 4; // N f32
static const size_t OFF_GN_SUM   = OFF_NODE_CNT + (size_t)NN * 4;        // G*128 f32
static const size_t OFF_GN_CNT   = OFF_GN_SUM + (size_t)GG * UDIM * 4;   // G f32
static const size_t OFF_GE_SUM   = OFF_GN_CNT + (size_t)GG * 4;          // G*128 f32
static const size_t OFF_GE_CNT   = OFF_GE_SUM + (size_t)GG * UDIM * 4;   // G f32
static const size_t WS_END       = OFF_GE_CNT + (size_t)GG * 4;

// ---------------- small utility kernels ----------------
__global__ void zero_kernel(float* p, long n) {
  long i = (long)blockIdx.x * blockDim.x + threadIdx.x;
  long stride = (long)gridDim.x * blockDim.x;
  for (; i < n; i += stride) p[i] = 0.f;
}

// out[col*K + k] = (bf16) W[k*128 + col]   (W is K x 128 row-major fp32)
__global__ void wconv_kernel(const float* __restrict__ W, __bf16* __restrict__ out, int K) {
  int idx = blockIdx.x * blockDim.x + threadIdx.x;
  if (idx >= K * 128) return;
  int col = idx & 127;
  int k = idx >> 7;
  out[(size_t)col * K + k] = (__bf16)W[(size_t)k * 128 + col];
}

__global__ void count_kernel(const int* __restrict__ eidx, const int* __restrict__ gidx,
                             float* node_cnt, float* ge_cnt, float* gn_cnt) {
  long i = (long)blockIdx.x * blockDim.x + threadIdx.x;
  if (i < EE) {
    int s = eidx[i * 2];
    atomicAdd(&node_cnt[s], 1.f);
    atomicAdd(&ge_cnt[gidx[s]], 1.f);
  } else if (i < (long)EE + NN) {
    int n = (int)(i - EE);
    atomicAdd(&gn_cnt[gidx[n]], 1.f);
  }
}

// ---------------- WMMA fragment loader ----------------
// 16-bit A-layout fragment for a 16x32 tile in LDS (row-major, stride in
// halfwords). lanes 0-15: row=lane, K 0..7 & 16..23 ; lanes 16-31: row=lane-16,
// K 8..15 & 24..31.  B fragments use the same pattern on the [col][K] weight
// buffer (B^T stored row-major).
__device__ __forceinline__ v16bf load_frag(const __bf16* base, int stride, int lane) {
  const int r = lane & 15;
  const __bf16* p = base + r * stride + ((lane >> 4) & 1) * 8;
  v16bf f;
#pragma unroll
  for (int i = 0; i < 8; ++i) { f[i] = p[i]; f[i + 8] = p[i + 16]; }
  return f;
}

__device__ __forceinline__ v8f vzero8() {
  v8f z = {0.f, 0.f, 0.f, 0.f, 0.f, 0.f, 0.f, 0.f};
  return z;
}

// ---------------- weight tile staging: TDM (tensor_load_to_lds) ----------------
// Stages a [128 cols][32 k] bf16 tile from w (row stride K elements, starting at
// k-offset kb) into Wbuf.  D# built per CDNA5 ISA:
//   group0: count=1, lds_addr, 57b global_addr, type=2
//   group1: data_size=1(2B), tensor_dim0=K, tensor_dim1=128, tile_dim0=32,
//           tile_dim1=128, tensor_dim0_stride=K
// 6-arg builtin form (clang-23 / therock-10.0 headers): extra int32x8 group
// before cpol; zero-filled (only meaningful for >2D tensors / gather mode).
// Caller must __syncthreads() afterwards.
__device__ __forceinline__ void stage_weights(const __bf16* __restrict__ w, int K, int kb,
                                              __bf16* Wbuf, int tid, int wave) {
#ifdef HAVE_TDM
  if (wave == 0) {
    unsigned lds = (unsigned)(uintptr_t)(void*)Wbuf;   // low 32 bits = LDS offset
    unsigned long long ga =
        (unsigned long long)(uintptr_t)(const void*)w + (unsigned long long)kb * 2ull;
    v4u g0;
    g0[0] = 1u;                                        // count=1 (valid descriptor)
    g0[1] = lds;                                       // lds_addr
    g0[2] = (unsigned)ga;                              // global_addr[31:0]
    g0[3] = (unsigned)(ga >> 32) | 0x80000000u;        // global_addr[56:32] | type=2
    v8i_ g1;
    g1[0] = 0x00010000;                                // data_size=1 (2 bytes)
    g1[1] = (int)((unsigned)(K & 0xFFFF) << 16);       // tensor_dim0[15:0] -> bits 63:48
    g1[2] = (int)(128u << 16);                         // tensor_dim0[31:16]=0 | tensor_dim1=128
    g1[3] = (int)(32u << 16);                          // tensor_dim1[31:16]=0 | tile_dim0=32
    g1[4] = 128;                                       // tile_dim1=128, tile_dim2=0
    g1[5] = K;                                         // tensor_dim0_stride[31:0]
    g1[6] = 0;                                         // stride hi / dim1_stride lo
    g1[7] = 0;
    v4i_ z4; z4[0] = 0; z4[1] = 0; z4[2] = 0; z4[3] = 0;   // groups 2/3 unused (2D)
    v8i_ z8;
#pragma unroll
    for (int i = 0; i < 8; ++i) z8[i] = 0;
    __builtin_amdgcn_tensor_load_to_lds(g0, g1, z4, z4, z8, 0);
    __builtin_amdgcn_s_wait_tensorcnt(0);
  }
#else
  // fallback: vectorized LDS copy (16B chunks)
#pragma unroll
  for (int j = 0; j < 2; ++j) {
    int q = tid + j * 256;          // 512 x 16B chunks = 8KB
    int col = q >> 2;
    int cc = (q & 3) * 8;
    *(int4*)&Wbuf[col * 32 + cc] = *(const int4*)&w[(size_t)col * K + kb + cc];
  }
#endif
}

// ---------------- edge MLP: K=512 -> 128 -> 128 ----------------
__global__ __launch_bounds__(256) void edge_mlp_kernel(
    const float* __restrict__ nodef, const float* __restrict__ edgef,
    const float* __restrict__ globf, const int* __restrict__ eidx,
    const int* __restrict__ gidx,
    const __bf16* __restrict__ w1, const __bf16* __restrict__ w2,
    const float* __restrict__ s1, const float* __restrict__ sh1,
    const float* __restrict__ s2, const float* __restrict__ sh2,
    float* __restrict__ edge_out, float* __restrict__ node_sum,
    float* __restrict__ ge_sum) {
  __shared__ __bf16 Abuf[128 * 32];
  __shared__ __bf16 Wbuf[128 * 32];
  __shared__ __bf16 Hbuf[128 * 128];
  __shared__ int sRow[4][128];   // 0:src 1:dst 2:self(edge row) 3:edge-graph

  const int tid = threadIdx.x, lane = tid & 31, wave = tid >> 5;
  const long base = (long)blockIdx.x * 128;

  if (tid < 128) {
    int s = eidx[(base + tid) * 2 + 0];
    int d = eidx[(base + tid) * 2 + 1];
    sRow[0][tid] = s; sRow[1][tid] = d;
    sRow[2][tid] = (int)(base + tid);
    sRow[3][tid] = gidx[s];
  }
  __syncthreads();

  v8f acc[8];
#pragma unroll
  for (int t = 0; t < 8; ++t) acc[t] = vzero8();

  // ---- layer 1: K = 512 (16 k-steps of 32); segment select is uniform per kc ----
  for (int kc = 0; kc < 16; ++kc) {
    const int seg = kc >> 2;
    const int kb = (kc & 3) * 32;
    const float* P = (seg <= 1) ? nodef : (seg == 2 ? edgef : globf);
    const int* R = sRow[seg];
#pragma unroll
    for (int j = 0; j < 4; ++j) {       // 128 rows x 32 k as float4 loads
      int q = tid + j * 256;
      int r = q >> 3;
      int c = (q & 7) * 4;
      const float4 vv = *(const float4*)&P[(long)R[r] * UDIM + kb + c];
      __bf16* dp = &Abuf[r * 32 + c];
      dp[0] = (__bf16)vv.x; dp[1] = (__bf16)vv.y;
      dp[2] = (__bf16)vv.z; dp[3] = (__bf16)vv.w;
    }
    stage_weights(w1, 512, kc * 32, Wbuf, tid, wave);
    __syncthreads();
    v16bf a = load_frag(&Abuf[wave * 16 * 32], 32, lane);
#pragma unroll
    for (int t = 0; t < 8; ++t) {
      v16bf b = load_frag(&Wbuf[t * 16 * 32], 32, lane);
      acc[t] = __builtin_amdgcn_wmma_f32_16x16x32_bf16(false, a, false, b,
                                                       (short)0, acc[t], false, false);
    }
    __syncthreads();
  }

  // ---- epilogue 1: scale/shift/relu -> Hbuf bf16 ----
#pragma unroll
  for (int t = 0; t < 8; ++t) {
    int col = t * 16 + (lane & 15);
    int rbase = wave * 16 + ((lane >> 4) & 1) * 8;
    float sc = s1[col], sh = sh1[col];
#pragma unroll
    for (int g = 0; g < 8; ++g) {
      float v = acc[t][g] * sc + sh;
      v = v > 0.f ? v : 0.f;
      Hbuf[(rbase + g) * 128 + col] = (__bf16)v;
    }
  }
  __syncthreads();

  // ---- layer 2: K = 128 (4 k-steps) ----
#pragma unroll
  for (int t = 0; t < 8; ++t) acc[t] = vzero8();
  for (int kc = 0; kc < 4; ++kc) {
    stage_weights(w2, 128, kc * 32, Wbuf, tid, wave);
    __syncthreads();
    v16bf a = load_frag(&Hbuf[wave * 16 * 128 + kc * 32], 128, lane);
#pragma unroll
    for (int t = 0; t < 8; ++t) {
      v16bf b = load_frag(&Wbuf[t * 16 * 32], 32, lane);
      acc[t] = __builtin_amdgcn_wmma_f32_16x16x32_bf16(false, a, false, b,
                                                       (short)0, acc[t], false, false);
    }
    __syncthreads();
  }

  // ---- epilogue 2: store edge_upd + segment-sum atomics ----
#pragma unroll
  for (int t = 0; t < 8; ++t) {
    int col = t * 16 + (lane & 15);
    int rbase = wave * 16 + ((lane >> 4) & 1) * 8;
    float sc = s2[col], sh = sh2[col];
#pragma unroll
    for (int g = 0; g < 8; ++g) {
      int r = rbase + g;
      float v = acc[t][g] * sc + sh;
      v = v > 0.f ? v : 0.f;
      edge_out[(base + r) * UDIM + col] = v;
      atomicAdd(&node_sum[(long)sRow[0][r] * UDIM + col], v);
      atomicAdd(&ge_sum[(long)sRow[3][r] * UDIM + col], v);
    }
  }
}

// ---------------- node MLP: K=384 -> 128 -> 128 ----------------
__global__ __launch_bounds__(256) void node_mlp_kernel(
    const float* __restrict__ nodef, const float* __restrict__ globf,
    const int* __restrict__ gidx, const float* __restrict__ node_sum,
    const float* __restrict__ node_cnt,
    const __bf16* __restrict__ w1, const __bf16* __restrict__ w2,
    const float* __restrict__ s1, const float* __restrict__ sh1,
    const float* __restrict__ s2, const float* __restrict__ sh2,
    float* __restrict__ node_out, float* __restrict__ gn_sum) {
  __shared__ __bf16 Abuf[128 * 32];
  __shared__ __bf16 Wbuf[128 * 32];
  __shared__ __bf16 Hbuf[128 * 128];
  __shared__ int sGid[128];
  __shared__ float sRcp[128];

  const int tid = threadIdx.x, lane = tid & 31, wave = tid >> 5;
  const long base = (long)blockIdx.x * 128;

  if (tid < 128) {
    long grow = base + tid;
    if (grow < NN) {
      sGid[tid] = gidx[grow];
      float c = node_cnt[grow];
      sRcp[tid] = 1.f / (c > 1.f ? c : 1.f);
    } else { sGid[tid] = 0; sRcp[tid] = 0.f; }
  }
  __syncthreads();

  v8f acc[8];
#pragma unroll
  for (int t = 0; t < 8; ++t) acc[t] = vzero8();

  for (int kc = 0; kc < 12; ++kc) {       // K = 384, seg uniform per kc
    const int seg = kc >> 2;
    const int kb = (kc & 3) * 32;
#pragma unroll
    for (int j = 0; j < 4; ++j) {
      int q = tid + j * 256;
      int r = q >> 3;
      int c = (q & 7) * 4;
      long grow = base + r;
      float4 vv = make_float4(0.f, 0.f, 0.f, 0.f);
      if (grow < NN) {
        if (seg == 0) {
          vv = *(const float4*)&nodef[grow * UDIM + kb + c];
        } else if (seg == 1) {
          vv = *(const float4*)&node_sum[grow * UDIM + kb + c];
          float rc = sRcp[r];
          vv.x *= rc; vv.y *= rc; vv.z *= rc; vv.w *= rc;
        } else {
          vv = *(const float4*)&globf[(long)sGid[r] * UDIM + kb + c];
        }
      }
      __bf16* dp = &Abuf[r * 32 + c];
      dp[0] = (__bf16)vv.x; dp[1] = (__bf16)vv.y;
      dp[2] = (__bf16)vv.z; dp[3] = (__bf16)vv.w;
    }
    stage_weights(w1, 384, kc * 32, Wbuf, tid, wave);
    __syncthreads();
    v16bf a = load_frag(&Abuf[wave * 16 * 32], 32, lane);
#pragma unroll
    for (int t = 0; t < 8; ++t) {
      v16bf b = load_frag(&Wbuf[t * 16 * 32], 32, lane);
      acc[t] = __builtin_amdgcn_wmma_f32_16x16x32_bf16(false, a, false, b,
                                                       (short)0, acc[t], false, false);
    }
    __syncthreads();
  }

#pragma unroll
  for (int t = 0; t < 8; ++t) {
    int col = t * 16 + (lane & 15);
    int rbase = wave * 16 + ((lane >> 4) & 1) * 8;
    float sc = s1[col], sh = sh1[col];
#pragma unroll
    for (int g = 0; g < 8; ++g) {
      float v = acc[t][g] * sc + sh;
      v = v > 0.f ? v : 0.f;
      Hbuf[(rbase + g) * 128 + col] = (__bf16)v;
    }
  }
  __syncthreads();

#pragma unroll
  for (int t = 0; t < 8; ++t) acc[t] = vzero8();
  for (int kc = 0; kc < 4; ++kc) {        // K = 128
    stage_weights(w2, 128, kc * 32, Wbuf, tid, wave);
    __syncthreads();
    v16bf a = load_frag(&Hbuf[wave * 16 * 128 + kc * 32], 128, lane);
#pragma unroll
    for (int t = 0; t < 8; ++t) {
      v16bf b = load_frag(&Wbuf[t * 16 * 32], 32, lane);
      acc[t] = __builtin_amdgcn_wmma_f32_16x16x32_bf16(false, a, false, b,
                                                       (short)0, acc[t], false, false);
    }
    __syncthreads();
  }

#pragma unroll
  for (int t = 0; t < 8; ++t) {
    int col = t * 16 + (lane & 15);
    int rbase = wave * 16 + ((lane >> 4) & 1) * 8;
    float sc = s2[col], sh = sh2[col];
#pragma unroll
    for (int g = 0; g < 8; ++g) {
      int r = rbase + g;
      long grow = base + r;
      if (grow < NN) {
        float v = acc[t][g] * sc + sh;
        v = v > 0.f ? v : 0.f;
        node_out[grow * UDIM + col] = v;
        atomicAdd(&gn_sum[(long)sGid[r] * UDIM + col], v);
      }
    }
  }
}

// ---------------- global MLP: K=384 -> 128 -> 128 (G=256, 2 blocks) ----------------
__global__ __launch_bounds__(256) void global_mlp_kernel(
    const float* __restrict__ globf, const float* __restrict__ gn_sum,
    const float* __restrict__ gn_cnt, const float* __restrict__ ge_sum,
    const float* __restrict__ ge_cnt,
    const __bf16* __restrict__ w1, const __bf16* __restrict__ w2,
    const float* __restrict__ s1, const float* __restrict__ sh1,
    const float* __restrict__ s2, const float* __restrict__ sh2,
    float* __restrict__ glob_out) {
  __shared__ __bf16 Abuf[128 * 32];
  __shared__ __bf16 Wbuf[128 * 32];
  __shared__ __bf16 Hbuf[128 * 128];
  __shared__ float sRcpN[128], sRcpE[128];

  const int tid = threadIdx.x, lane = tid & 31, wave = tid >> 5;
  const long base = (long)blockIdx.x * 128;

  if (tid < 128) {
    float cn = gn_cnt[base + tid];
    float ce = ge_cnt[base + tid];
    sRcpN[tid] = 1.f / (cn > 1.f ? cn : 1.f);
    sRcpE[tid] = 1.f / (ce > 1.f ? ce : 1.f);
  }
  __syncthreads();

  v8f acc[8];
#pragma unroll
  for (int t = 0; t < 8; ++t) acc[t] = vzero8();

  for (int kc = 0; kc < 12; ++kc) {       // K = 384
    const int seg = kc >> 2;
    const int kb = (kc & 3) * 32;
#pragma unroll
    for (int j = 0; j < 4; ++j) {
      int q = tid + j * 256;
      int r = q >> 3;
      int c = (q & 7) * 4;
      long grow = base + r;
      float4 vv;
      if (seg == 0) {
        vv = *(const float4*)&gn_sum[grow * UDIM + kb + c];
        float rc = sRcpN[r];
        vv.x *= rc; vv.y *= rc; vv.z *= rc; vv.w *= rc;
      } else if (seg == 1) {
        vv = *(const float4*)&ge_sum[grow * UDIM + kb + c];
        float rc = sRcpE[r];
        vv.x *= rc; vv.y *= rc; vv.z *= rc; vv.w *= rc;
      } else {
        vv = *(const float4*)&globf[grow * UDIM + kb + c];
      }
      __bf16* dp = &Abuf[r * 32 + c];
      dp[0] = (__bf16)vv.x; dp[1] = (__bf16)vv.y;
      dp[2] = (__bf16)vv.z; dp[3] = (__bf16)vv.w;
    }
    stage_weights(w1, 384, kc * 32, Wbuf, tid, wave);
    __syncthreads();
    v16bf a = load_frag(&Abuf[wave * 16 * 32], 32, lane);
#pragma unroll
    for (int t = 0; t < 8; ++t) {
      v16bf b = load_frag(&Wbuf[t * 16 * 32], 32, lane);
      acc[t] = __builtin_amdgcn_wmma_f32_16x16x32_bf16(false, a, false, b,
                                                       (short)0, acc[t], false, false);
    }
    __syncthreads();
  }

#pragma unroll
  for (int t = 0; t < 8; ++t) {
    int col = t * 16 + (lane & 15);
    int rbase = wave * 16 + ((lane >> 4) & 1) * 8;
    float sc = s1[col], sh = sh1[col];
#pragma unroll
    for (int g = 0; g < 8; ++g) {
      float v = acc[t][g] * sc + sh;
      v = v > 0.f ? v : 0.f;
      Hbuf[(rbase + g) * 128 + col] = (__bf16)v;
    }
  }
  __syncthreads();

#pragma unroll
  for (int t = 0; t < 8; ++t) acc[t] = vzero8();
  for (int kc = 0; kc < 4; ++kc) {        // K = 128
    stage_weights(w2, 128, kc * 32, Wbuf, tid, wave);
    __syncthreads();
    v16bf a = load_frag(&Hbuf[wave * 16 * 128 + kc * 32], 128, lane);
#pragma unroll
    for (int t = 0; t < 8; ++t) {
      v16bf b = load_frag(&Wbuf[t * 16 * 32], 32, lane);
      acc[t] = __builtin_amdgcn_wmma_f32_16x16x32_bf16(false, a, false, b,
                                                       (short)0, acc[t], false, false);
    }
    __syncthreads();
  }

#pragma unroll
  for (int t = 0; t < 8; ++t) {
    int col = t * 16 + (lane & 15);
    int rbase = wave * 16 + ((lane >> 4) & 1) * 8;
    float sc = s2[col], sh = sh2[col];
#pragma unroll
    for (int g = 0; g < 8; ++g) {
      float v = acc[t][g] * sc + sh;
      v = v > 0.f ? v : 0.f;
      glob_out[(base + rbase + g) * UDIM + col] = v;
    }
  }
}

// ---------------- launcher ----------------
extern "C" void kernel_launch(void* const* d_in, const int* in_sizes, int n_in,
                              void* d_out, int out_size, void* d_ws, size_t ws_size,
                              hipStream_t stream) {
  const float* nodef = (const float*)d_in[0];
  const float* edgef = (const float*)d_in[1];
  const float* globf = (const float*)d_in[2];
  const int*   eidx  = (const int*)d_in[3];
  const int*   gidx  = (const int*)d_in[4];
  const float* e_W1 = (const float*)d_in[5];
  const float* e_s1 = (const float*)d_in[6];
  const float* e_sh1 = (const float*)d_in[7];
  const float* e_W2 = (const float*)d_in[8];
  const float* e_s2 = (const float*)d_in[9];
  const float* e_sh2 = (const float*)d_in[10];
  const float* n_W1 = (const float*)d_in[11];
  const float* n_s1 = (const float*)d_in[12];
  const float* n_sh1 = (const float*)d_in[13];
  const float* n_W2 = (const float*)d_in[14];
  const float* n_s2 = (const float*)d_in[15];
  const float* n_sh2 = (const float*)d_in[16];
  const float* g_W1 = (const float*)d_in[17];
  const float* g_s1 = (const float*)d_in[18];
  const float* g_sh1 = (const float*)d_in[19];
  const float* g_W2 = (const float*)d_in[20];
  const float* g_s2 = (const float*)d_in[21];
  const float* g_sh2 = (const float*)d_in[22];

  char* ws = (char*)d_ws;
  __bf16* we1 = (__bf16*)(ws + OFF_WE1);
  __bf16* we2 = (__bf16*)(ws + OFF_WE2);
  __bf16* wn1 = (__bf16*)(ws + OFF_WN1);
  __bf16* wn2 = (__bf16*)(ws + OFF_WN2);
  __bf16* wg1 = (__bf16*)(ws + OFF_WG1);
  __bf16* wg2 = (__bf16*)(ws + OFF_WG2);
  float* node_sum = (float*)(ws + OFF_NODE_SUM);
  float* node_cnt = (float*)(ws + OFF_NODE_CNT);
  float* gn_sum = (float*)(ws + OFF_GN_SUM);
  float* gn_cnt = (float*)(ws + OFF_GN_CNT);
  float* ge_sum = (float*)(ws + OFF_GE_SUM);
  float* ge_cnt = (float*)(ws + OFF_GE_CNT);

  float* out_node = (float*)d_out;                      // N*128
  float* out_edge = out_node + (size_t)NN * UDIM;       // E*128
  float* out_glob = out_edge + (size_t)EE * UDIM;       // G*128

  // 1) zero accumulators & counts
  long nzero = (long)((WS_END - OFF_NODE_SUM) / 4);
  zero_kernel<<<4096, 256, 0, stream>>>(node_sum, nzero);

  // 2) weight transposition / bf16 conversion
  wconv_kernel<<<(512 * 128 + 255) / 256, 256, 0, stream>>>(e_W1, we1, 512);
  wconv_kernel<<<(128 * 128 + 255) / 256, 256, 0, stream>>>(e_W2, we2, 128);
  wconv_kernel<<<(384 * 128 + 255) / 256, 256, 0, stream>>>(n_W1, wn1, 384);
  wconv_kernel<<<(128 * 128 + 255) / 256, 256, 0, stream>>>(n_W2, wn2, 128);
  wconv_kernel<<<(384 * 128 + 255) / 256, 256, 0, stream>>>(g_W1, wg1, 384);
  wconv_kernel<<<(128 * 128 + 255) / 256, 256, 0, stream>>>(g_W2, wg2, 128);

  // 3) segment counts
  count_kernel<<<(EE + NN + 255) / 256, 256, 0, stream>>>(eidx, gidx, node_cnt, ge_cnt, gn_cnt);

  // 4) edge MLP (+ node/graph edge sums)
  edge_mlp_kernel<<<EE / 128, 256, 0, stream>>>(nodef, edgef, globf, eidx, gidx,
                                                we1, we2, e_s1, e_sh1, e_s2, e_sh2,
                                                out_edge, node_sum, ge_sum);

  // 5) node MLP (+ graph node sums)
  node_mlp_kernel<<<(NN + 127) / 128, 256, 0, stream>>>(nodef, globf, gidx,
                                                        node_sum, node_cnt,
                                                        wn1, wn2, n_s1, n_sh1, n_s2, n_sh2,
                                                        out_node, gn_sum);

  // 6) global MLP
  global_mlp_kernel<<<GG / 128, 256, 0, stream>>>(globf, gn_sum, gn_cnt, ge_sum, ge_cnt,
                                                  wg1, wg2, g_s1, g_sh1, g_s2, g_sh2,
                                                  out_glob);
}
